// AttentionLSTM_88347477279632
// MI455X (gfx1250) — matrix operand order repaired and checked by reading, not compiled, not run on hardware
//
#include <hip/hip_runtime.h>
#include <hip/hip_bf16.h>
#include <math.h>

// ---------------------------------------------------------------------------
// CDNA5 (gfx1250) wave32 WMMA types
// ---------------------------------------------------------------------------
typedef __attribute__((ext_vector_type(16))) __bf16 v16bf;
typedef __attribute__((ext_vector_type(8)))  __bf16 v8bf;
typedef __attribute__((ext_vector_type(8)))  float  v8f;

#define DEV __device__ __forceinline__

DEV __bf16 f2bf(float f) {
  unsigned int u; __builtin_memcpy(&u, &f, 4);
  unsigned int r = (u + 0x7FFFu + ((u >> 16) & 1u)) >> 16;   // RNE
  unsigned short s = (unsigned short)r;
  __bf16 b; __builtin_memcpy(&b, &s, 2); return b;
}
DEV float bf2f(__bf16 b) {
  unsigned short s; __builtin_memcpy(&s, &b, 2);
  unsigned int u = ((unsigned int)s) << 16;
  float f; __builtin_memcpy(&f, &u, 4); return f;
}
DEV float sigmoidf(float x) { return 1.0f / (1.0f + __expf(-x)); }

DEV v8f zero8() {
  v8f z;
#pragma unroll
  for (int i = 0; i < 8; ++i) z[i] = 0.0f;
  return z;
}

// A fragment: 16x32 bf16, row-major source A[M][K] (lda=K).
// lanes 0-15: row=lane,  K = k0+0..7 (v0..3) and k0+16..23 (v4..7)
// lanes16-31: row=lane-16, K = k0+8..15 and k0+24..31
DEV v16bf load_a(const __bf16* __restrict__ A, int lda, int m0, int k0, int lane) {
  int r  = m0 + (lane & 15);
  int kb = k0 + ((lane >> 4) << 3);
  const __bf16* p = A + (size_t)r * lda + kb;
  v8bf lo = *reinterpret_cast<const v8bf*>(p);
  v8bf hi = *reinterpret_cast<const v8bf*>(p + 16);
  v16bf o;
#pragma unroll
  for (int i = 0; i < 8; ++i) { o[i] = lo[i]; o[i + 8] = hi[i]; }
  return o;
}
// B fragment: 32x16 bf16 tile of W^T, source row-major W[N][K] (ldw=K).
// lane: n = n0 + lane%16; lanes 0-15: K = k0+0..7 and k0+16..23; lanes 16-31: K = k0+8..15 and k0+24..31
// => two 8-element (16B) loads from one row of W.
DEV v16bf load_b(const __bf16* __restrict__ W, int ldw, int n0, int k0, int lane) {
  int n  = n0 + (lane & 15);
  int kb = k0 + ((lane >> 4) << 3);
  const __bf16* p = W + (size_t)n * ldw + kb;
  v8bf lo = *reinterpret_cast<const v8bf*>(p);
  v8bf hi = *reinterpret_cast<const v8bf*>(p + 16);
  v16bf o;
#pragma unroll
  for (int i = 0; i < 8; ++i) { o[i] = lo[i]; o[i + 8] = hi[i]; }
  return o;
}

// ---------------------------------------------------------------------------
// Utility kernels
// ---------------------------------------------------------------------------
__global__ void cast_f32_bf16(const float* __restrict__ s, __bf16* __restrict__ d, size_t n) {
  size_t i = (size_t)blockIdx.x * blockDim.x + threadIdx.x;
  size_t stride = (size_t)gridDim.x * blockDim.x;
  for (; i < n; i += stride) d[i] = f2bf(s[i]);
}
__global__ void zero_u32(unsigned int* __restrict__ p, size_t n) {
  size_t i = (size_t)blockIdx.x * blockDim.x + threadIdx.x;
  size_t stride = (size_t)gridDim.x * blockDim.x;
  for (; i < n; i += stride) p[i] = 0u;
}
// Xemb[(t*B+b)*E + e] = bf16(embed_w[x[b*T+t]*E + e])
__global__ void embed_gather(const int* __restrict__ x, const float* __restrict__ ew,
                             __bf16* __restrict__ X, int B, int T, int E) {
  size_t gid = (size_t)blockIdx.x * blockDim.x + threadIdx.x;
  size_t total = (size_t)B * T * E;
  if (gid >= total) return;
  int e = (int)(gid % E);
  size_t r = gid / E;
  int b = (int)(r % B), t = (int)(r / B);
  int tok = x[(size_t)b * T + t];
  X[gid] = f2bf(ew[(size_t)tok * E + e]);
}

// ---------------------------------------------------------------------------
// Big WMMA GEMM: C[M][N] = A[M][K] * W[N][K]^T + bias0 + bias1
// One wave computes a 16x(16*NT) strip, reusing the A fragment NT times.
// NT independent accumulators keep the WMMA pipe full.
// ---------------------------------------------------------------------------
template <int NT, bool OUT_BF16>
__global__ void wmma_gemm_bias(const __bf16* __restrict__ A, const __bf16* __restrict__ W,
                               const float* __restrict__ bias0, const float* __restrict__ bias1,
                               void* __restrict__ Cout, int M, int N, int K) {
  int lane = threadIdx.x & 31;
  int wave = (int)((blockIdx.x * blockDim.x + threadIdx.x) >> 5);
  int stripsN = N / (16 * NT);
  int m0 = (wave / stripsN) << 4;
  int nbase = (wave % stripsN) * (16 * NT);
  if (m0 >= M) return;
  v8f acc[NT];
#pragma unroll
  for (int j = 0; j < NT; ++j) acc[j] = zero8();
  for (int k0 = 0; k0 < K; k0 += 32) {
    v16bf a = load_a(A, K, m0, k0, lane);
#pragma unroll
    for (int j = 0; j < NT; ++j) {
      v16bf b = load_b(W, K, nbase + 16 * j, k0, lane);
      acc[j] = __builtin_amdgcn_wmma_f32_16x16x32_bf16(false, a, false, b,
                                                       (short)0, acc[j], false, false);
    }
  }
  int mrow = m0 + ((lane >> 4) << 3);
#pragma unroll
  for (int j = 0; j < NT; ++j) {
    int n = nbase + 16 * j + (lane & 15);
    float bias = 0.0f;
    if (bias0) bias += bias0[n];
    if (bias1) bias += bias1[n];
#pragma unroll
    for (int i = 0; i < 8; ++i) {
      float val = acc[j][i] + bias;
      if constexpr (OUT_BF16) ((__bf16*)Cout)[(size_t)(mrow + i) * N + n] = f2bf(val);
      else                    ((float*)Cout)[(size_t)(mrow + i) * N + n] = val;
    }
  }
}

// ---------------------------------------------------------------------------
// Recurrent-step gates GEMM: gates[b][n] = xproj_t[b][n] + h_bf[b][:] . Whh[n][:]
// M=64, N=4096, K=1024 -> 1024 waves, 1 tile per wave (maximize parallelism).
// Two independent accumulator chains (even/odd K blocks) halve the serial
// WMMA RAW-dependency latency on the time-critical recurrence.
// ---------------------------------------------------------------------------
__global__ void wmma_lstm_gates(const __bf16* __restrict__ hbf, const __bf16* __restrict__ Whh,
                                const __bf16* __restrict__ xrow, float* __restrict__ gates,
                                int N, int K) {
  int lane = threadIdx.x & 31;
  int wave = (int)((blockIdx.x * blockDim.x + threadIdx.x) >> 5);
  int tilesN = N >> 4;
  int m0 = (wave / tilesN) << 4;
  int n0 = (wave % tilesN) << 4;
  v8f acc0 = zero8();
  v8f acc1 = zero8();
  for (int k0 = 0; k0 < K; k0 += 64) {
    v16bf a0 = load_a(hbf, K, m0, k0, lane);
    v16bf b0 = load_b(Whh, K, n0, k0, lane);
    v16bf a1 = load_a(hbf, K, m0, k0 + 32, lane);
    v16bf b1 = load_b(Whh, K, n0, k0 + 32, lane);
    acc0 = __builtin_amdgcn_wmma_f32_16x16x32_bf16(false, a0, false, b0,
                                                   (short)0, acc0, false, false);
    acc1 = __builtin_amdgcn_wmma_f32_16x16x32_bf16(false, a1, false, b1,
                                                   (short)0, acc1, false, false);
  }
  int n = n0 + (lane & 15);
  int mrow = m0 + ((lane >> 4) << 3);
#pragma unroll
  for (int i = 0; i < 8; ++i) {
    size_t idx = (size_t)(mrow + i) * N + n;
    gates[idx] = acc0[i] + acc1[i] + bf2f(xrow[idx]);
  }
}

// gate nonlinearity + state update; optional layer-output writes
__global__ void lstm_pointwise(const float* __restrict__ gates, float* __restrict__ c,
                               __bf16* __restrict__ hbf,
                               __bf16* __restrict__ out_tb,      // layer0: [(t*B+b)*H+j] bf16
                               float* __restrict__ out_bt_f,     // layer1: [(b*T+t)*H+j] f32
                               __bf16* __restrict__ out_bt_bf,   // layer1: bf16 twin
                               int t, int B, int T, int H) {
  int gid = blockIdx.x * blockDim.x + threadIdx.x;
  if (gid >= B * H) return;
  int b = gid / H, j = gid % H;
  const float* g = gates + (size_t)b * 4 * H;
  float ig = sigmoidf(g[j]);
  float fg = sigmoidf(g[H + j]);
  float gg = tanhf(g[2 * H + j]);
  float og = sigmoidf(g[3 * H + j]);
  float cn = fg * c[gid] + ig * gg;
  float hn = og * tanhf(cn);
  c[gid] = cn;
  hbf[gid] = f2bf(hn);
  if (out_tb) out_tb[((size_t)t * B + b) * H + j] = f2bf(hn);
  if (out_bt_f) {
    size_t r = ((size_t)b * T + t) * H + j;
    out_bt_f[r] = hn;
    out_bt_bf[r] = f2bf(hn);
  }
}

// ---------------------------------------------------------------------------
// Attention + head (small, bandwidth-trivial)
// ---------------------------------------------------------------------------
__global__ void dense_m2(const float* __restrict__ out2_f, const float* __restrict__ m2_w,
                         const float* __restrict__ m2_b, float* __restrict__ m2,
                         int B, int T, int H, int A) {
  int gid = blockIdx.x * blockDim.x + threadIdx.x;
  if (gid >= B * A) return;
  int b = gid / A, a = gid % A;
  const float* hl = out2_f + ((size_t)b * T + (T - 1)) * H;
  const float* w = m2_w + (size_t)a * H;
  float s = 0.0f;
  for (int k = 0; k < H; ++k) s += hl[k] * w[k];
  m2[gid] = s + m2_b[a];
}

__global__ void attn_scores(const float* __restrict__ m1, const float* __restrict__ m2,
                            const float* __restrict__ v, float* __restrict__ scores,
                            int T, int A) {
  __shared__ float red[256];
  int r = blockIdx.x;
  int b = r / T;
  float s = 0.0f;
  for (int a = threadIdx.x; a < A; a += blockDim.x)
    s += tanhf(m1[(size_t)r * A + a] + m2[(size_t)b * A + a]) * v[a];
  red[threadIdx.x] = s; __syncthreads();
  for (int off = blockDim.x >> 1; off > 0; off >>= 1) {
    if ((int)threadIdx.x < off) red[threadIdx.x] += red[threadIdx.x + off];
    __syncthreads();
  }
  if (threadIdx.x == 0) scores[r] = red[0];
}

__global__ void attn_softmax(const float* __restrict__ scores, float* __restrict__ attn, int T) {
  __shared__ float red[256];
  __shared__ float smax, ssum;
  int b = blockIdx.x;
  int tid = threadIdx.x;
  float x = (tid < T) ? scores[(size_t)b * T + tid] : -1e30f;
  red[tid] = x; __syncthreads();
  for (int off = blockDim.x >> 1; off > 0; off >>= 1) {
    if (tid < off) red[tid] = fmaxf(red[tid], red[tid + off]);
    __syncthreads();
  }
  if (tid == 0) smax = red[0];
  __syncthreads();
  float e = (tid < T) ? __expf(x - smax) : 0.0f;
  red[tid] = e; __syncthreads();
  for (int off = blockDim.x >> 1; off > 0; off >>= 1) {
    if (tid < off) red[tid] += red[tid + off];
    __syncthreads();
  }
  if (tid == 0) ssum = red[0];
  __syncthreads();
  if (tid < T) attn[(size_t)b * T + tid] = e / ssum;
}

__global__ void attn_context(const float* __restrict__ out2_f, const float* __restrict__ attn,
                             float* __restrict__ ctx, int B, int T, int H) {
  int gid = blockIdx.x * blockDim.x + threadIdx.x;
  if (gid >= B * H) return;
  int b = gid / H, j = gid % H;
  float s = 0.0f;
  for (int t = 0; t < T; ++t) s += out2_f[((size_t)b * T + t) * H + j] * attn[(size_t)b * T + t];
  ctx[gid] = s;
}

__global__ void dense_nvec(const float* __restrict__ ctx, const float* __restrict__ out2_f,
                           const float* __restrict__ n_w, const float* __restrict__ n_b,
                           float* __restrict__ nvec, int B, int T, int H) {
  int gid = blockIdx.x * blockDim.x + threadIdx.x;
  if (gid >= B * H) return;
  int b = gid / H, j = gid % H;
  const float* w = n_w + (size_t)j * 2 * H;
  const float* hl = out2_f + ((size_t)b * T + (T - 1)) * H;
  const float* cb = ctx + (size_t)b * H;
  float s = n_b[j];
  for (int k = 0; k < H; ++k) s += cb[k] * w[k];
  for (int k = 0; k < H; ++k) s += hl[k] * w[H + k];
  nvec[gid] = tanhf(s);
}

__global__ void dense_logit(const float* __restrict__ nvec, const float* __restrict__ out_w,
                            const float* __restrict__ out_b, float* __restrict__ out,
                            int H, int C) {
  __shared__ float red0[256];
  __shared__ float red1[256];
  int b = blockIdx.x;
  int tid = threadIdx.x;
  float s0 = 0.0f, s1 = 0.0f;
  for (int k = tid; k < H; k += blockDim.x) {
    float nv = nvec[(size_t)b * H + k];
    s0 += nv * out_w[k];
    s1 += nv * out_w[H + k];
  }
  red0[tid] = s0; red1[tid] = s1; __syncthreads();
  for (int off = blockDim.x >> 1; off > 0; off >>= 1) {
    if (tid < off) { red0[tid] += red0[tid + off]; red1[tid] += red1[tid + off]; }
    __syncthreads();
  }
  if (tid == 0) {
    out[(size_t)b * C + 0] = red0[0] + out_b[0];
    out[(size_t)b * C + 1] = red1[0] + out_b[1];
  }
}

// ---------------------------------------------------------------------------
// Host orchestration
// ---------------------------------------------------------------------------
extern "C" void kernel_launch(void* const* d_in, const int* in_sizes, int n_in,
                              void* d_out, int out_size, void* d_ws, size_t ws_size,
                              hipStream_t stream) {
  const int B = 64, T = 256, E = 512, H = 1024, A = 512, C = 2;
  const int G = 4 * H;            // 4096 gates
  const int M = B * T;            // 16384 rows

  const int*   x       = (const int*)d_in[0];
  const float* embed_w = (const float*)d_in[1];
  const float* w_ih0   = (const float*)d_in[2];
  const float* w_hh0   = (const float*)d_in[3];
  const float* b_ih0   = (const float*)d_in[4];
  const float* b_hh0   = (const float*)d_in[5];
  const float* w_ih1   = (const float*)d_in[6];
  const float* w_hh1   = (const float*)d_in[7];
  const float* b_ih1   = (const float*)d_in[8];
  const float* b_hh1   = (const float*)d_in[9];
  const float* m1_w    = (const float*)d_in[10];
  const float* m1_b    = (const float*)d_in[11];
  const float* m2_w    = (const float*)d_in[12];
  const float* m2_b    = (const float*)d_in[13];
  const float* vvec    = (const float*)d_in[14];
  const float* n_w     = (const float*)d_in[15];
  const float* n_b     = (const float*)d_in[16];
  const float* out_w   = (const float*)d_in[17];
  const float* out_b   = (const float*)d_in[18];

  char* ws = (char*)d_ws;
  size_t off = 0;
  auto alloc = [&](size_t bytes) -> char* {
    char* p = ws + off;
    off += (bytes + 255) & ~(size_t)255;
    return p;
  };
  __bf16* wb_ih0  = (__bf16*)alloc((size_t)G * E * 2);
  __bf16* wb_hh0  = (__bf16*)alloc((size_t)G * H * 2);
  __bf16* wb_ih1  = (__bf16*)alloc((size_t)G * H * 2);
  __bf16* wb_hh1  = (__bf16*)alloc((size_t)G * H * 2);
  __bf16* wm1     = (__bf16*)alloc((size_t)A * H * 2);
  __bf16* Xemb    = (__bf16*)alloc((size_t)M * E * 2);
  __bf16* xproj0  = (__bf16*)alloc((size_t)M * G * 2);
  __bf16* xproj1  = (__bf16*)alloc((size_t)M * G * 2);
  __bf16* out1_tb = (__bf16*)alloc((size_t)M * H * 2);
  float*  out2_f  = (float*) alloc((size_t)M * H * 4);
  __bf16* out2_bf = (__bf16*)alloc((size_t)M * H * 2);
  float*  c0      = (float*) alloc((size_t)B * H * 4);
  float*  c1      = (float*) alloc((size_t)B * H * 4);
  __bf16* hbf0    = (__bf16*)alloc((size_t)B * H * 2);
  __bf16* hbf1    = (__bf16*)alloc((size_t)B * H * 2);
  float*  gates   = (float*) alloc((size_t)B * G * 4);
  float*  m1buf   = (float*) alloc((size_t)M * A * 4);
  float*  m2buf   = (float*) alloc((size_t)B * A * 4);
  float*  scores  = (float*) alloc((size_t)B * T * 4);
  float*  attn    = (float*) alloc((size_t)B * T * 4);
  float*  ctx     = (float*) alloc((size_t)B * H * 4);
  float*  nvec    = (float*) alloc((size_t)B * H * 4);
  (void)ws_size; (void)in_sizes; (void)n_in; (void)out_size;

  auto gsz = [](size_t n) { size_t b = (n + 255) / 256; return (int)(b > 4096 ? 4096 : b); };

  // 1) weights -> bf16
  cast_f32_bf16<<<gsz((size_t)G * E), 256, 0, stream>>>(w_ih0, wb_ih0, (size_t)G * E);
  cast_f32_bf16<<<gsz((size_t)G * H), 256, 0, stream>>>(w_hh0, wb_hh0, (size_t)G * H);
  cast_f32_bf16<<<gsz((size_t)G * H), 256, 0, stream>>>(w_ih1, wb_ih1, (size_t)G * H);
  cast_f32_bf16<<<gsz((size_t)G * H), 256, 0, stream>>>(w_hh1, wb_hh1, (size_t)G * H);
  cast_f32_bf16<<<gsz((size_t)A * H), 256, 0, stream>>>(m1_w, wm1, (size_t)A * H);

  // 2) zero LSTM state
  zero_u32<<<gsz((size_t)B * H), 256, 0, stream>>>((unsigned int*)c0,   (size_t)B * H);
  zero_u32<<<gsz((size_t)B * H), 256, 0, stream>>>((unsigned int*)c1,   (size_t)B * H);
  zero_u32<<<gsz((size_t)B * H / 2), 256, 0, stream>>>((unsigned int*)hbf0, (size_t)B * H / 2);
  zero_u32<<<gsz((size_t)B * H / 2), 256, 0, stream>>>((unsigned int*)hbf1, (size_t)B * H / 2);

  // 3) embedding gather -> Xemb[(t*B+b)][e]
  {
    size_t tot = (size_t)M * E;
    embed_gather<<<(int)((tot + 255) / 256), 256, 0, stream>>>(x, embed_w, Xemb, B, T, E);
  }

  // 4) x_proj0 = Xemb @ w_ih0^T + b_ih0 + b_hh0   (M=16384,N=4096,K=512)
  {
    int waves = (M / 16) * (G / 64);
    wmma_gemm_bias<4, true><<<waves * 32 / 256, 256, 0, stream>>>(
        Xemb, wb_ih0, b_ih0, b_hh0, xproj0, M, G, E);
  }

  // 5) layer-0 recurrence
  {
    int gwaves = (B / 16) * (G / 16);            // 1024
    int gblocks = gwaves * 32 / 256;             // 128
    int pblocks = (B * H + 255) / 256;
    for (int t = 0; t < T; ++t) {
      wmma_lstm_gates<<<gblocks, 256, 0, stream>>>(
          hbf0, wb_hh0, xproj0 + (size_t)t * B * G, gates, G, H);
      lstm_pointwise<<<pblocks, 256, 0, stream>>>(
          gates, c0, hbf0, out1_tb, nullptr, nullptr, t, B, T, H);
    }
  }

  // 6) x_proj1 = out1 @ w_ih1^T + b_ih1 + b_hh1   (M=16384,N=4096,K=1024)
  {
    int waves = (M / 16) * (G / 64);
    wmma_gemm_bias<4, true><<<waves * 32 / 256, 256, 0, stream>>>(
        out1_tb, wb_ih1, b_ih1, b_hh1, xproj1, M, G, H);
  }

  // 7) layer-1 recurrence -> out2 (f32 + bf16, [b][t][h])
  {
    int gwaves = (B / 16) * (G / 16);
    int gblocks = gwaves * 32 / 256;
    int pblocks = (B * H + 255) / 256;
    for (int t = 0; t < T; ++t) {
      wmma_lstm_gates<<<gblocks, 256, 0, stream>>>(
          hbf1, wb_hh1, xproj1 + (size_t)t * B * G, gates, G, H);
      lstm_pointwise<<<pblocks, 256, 0, stream>>>(
          gates, c1, hbf1, nullptr, out2_f, out2_bf, t, B, T, H);
    }
  }

  // 8) m1 = out2 @ m1_w^T + m1_b  (M=16384,N=512,K=1024, f32 out)
  {
    int waves = (M / 16) * (A / 64);
    wmma_gemm_bias<4, false><<<waves * 32 / 256, 256, 0, stream>>>(
        out2_bf, wm1, m1_b, nullptr, m1buf, M, A, H);
  }

  // 9) attention + head
  dense_m2<<<(B * A + 255) / 256, 256, 0, stream>>>(out2_f, m2_w, m2_b, m2buf, B, T, H, A);
  attn_scores<<<B * T, 256, 0, stream>>>(m1buf, m2buf, vvec, scores, T, A);
  attn_softmax<<<B, 256, 0, stream>>>(scores, attn, T);
  attn_context<<<(B * H + 255) / 256, 256, 0, stream>>>(out2_f, attn, ctx, B, T, H);
  dense_nvec<<<(B * H + 255) / 256, 256, 0, stream>>>(ctx, out2_f, n_w, n_b, nvec, B, T, H);
  dense_logit<<<B, 256, 0, stream>>>(nvec, out_w, out_b, (float*)d_out, H, C);
}
